// MambaPose_6339371729458
// MI455X (gfx1250) — matrix operand order repaired and checked by reading, not compiled, not run on hardware
//
#include <hip/hip_runtime.h>

// ---------------- model constants ----------------
#define BATCHN 4
#define TFRM   243
#define JNT    17
#define LSEQ   8262      // 2 * 243 * 17 per batch
#define NPOS   16524     // B * T * J
#define NROWS  33048     // B * LSEQ (valid token rows)
#define MPAD   33152     // NROWS padded to multiple of 128 (259 * 128)
#define MPE    16640     // NPOS padded to multiple of 128 (130 * 128)
#define DMODEL 384
#define DINNER 768
#define DSTATE 16

typedef __attribute__((ext_vector_type(16))) __bf16 bf16x16;
typedef __attribute__((ext_vector_type(8)))  float  f32x8;

__device__ __forceinline__ unsigned short f2bf(float f) {
    unsigned int u = __float_as_uint(f);
    u += 0x7FFFu + ((u >> 16) & 1u);           // round-to-nearest-even
    return (unsigned short)(u >> 16);
}

// ======================================================================
// Generic WMMA GEMM:  C[M x N] = A[M x K] * W[N x K]^T
// A, W bf16 (as ushort), C f32.  M = gridDim.y*128 (pre-padded, no guards),
// N = gridDim.x*64, K multiple of 32.
// Block: 256 threads = 8 waves. Each wave owns a 16-row strip x 64 cols.
// Double-buffered LDS tiles filled by async global->LDS copies (ASYNCcnt),
// so the next K-step's tiles stream in while WMMAs consume the current one.
// ======================================================================
__global__ __launch_bounds__(256) void gemm_wmma(
    const unsigned short* __restrict__ A,
    const unsigned short* __restrict__ W,
    float* __restrict__ C, int K, int N)
{
    __shared__ __align__(16) unsigned short As[2][128 * 32];   // 2 x 8 KB
    __shared__ __align__(16) unsigned short Ws[2][64 * 32];    // 2 x 4 KB

    const int tid  = threadIdx.x;
    const int wave = tid >> 5;
    const int lane = tid & 31;
    const int l16  = lane & 15;
    const int hi   = lane >> 4;
    const long m0  = (long)blockIdx.y * 128;
    const int  n0  = blockIdx.x * 64;

    // ---- per-lane async-copy slice addressing (16B chunks) ----
    // A tile: 128 rows x 32 cols = 512 chunks; wave w -> chunks [w*64, w*64+64)
    // W tile: 64 rows x 32 cols  = 256 chunks; wave w -> chunks [w*32, w*32+32)
    const unsigned short* Abase = A + m0 * (long)K;
    const unsigned short* Wbase = W + (long)n0 * (long)K;
    int ca0 = wave * 64 + lane, ca1 = ca0 + 32, cw0 = wave * 32 + lane;
    int ra0 = ca0 >> 2, pa0 = ca0 & 3;
    int ra1 = ca1 >> 2, pa1 = ca1 & 3;
    int rw0 = cw0 >> 2, pw0 = cw0 & 3;
    unsigned voffA0 = (unsigned)(ra0 * K + pa0 * 8) * 2u;
    unsigned voffA1 = (unsigned)(ra1 * K + pa1 * 8) * 2u;
    unsigned voffW0 = (unsigned)(rw0 * K + pw0 * 8) * 2u;
    // low 32 bits of a generic address of a __shared__ object == LDS offset
    unsigned ldsA0[2], ldsA1[2], ldsW0[2];
#pragma unroll
    for (int b = 0; b < 2; ++b) {
        ldsA0[b] = (unsigned)(uintptr_t)&As[b][ra0 * 32 + pa0 * 8];
        ldsA1[b] = (unsigned)(uintptr_t)&As[b][ra1 * 32 + pa1 * 8];
        ldsW0[b] = (unsigned)(uintptr_t)&Ws[b][rw0 * 32 + pw0 * 8];
    }

    auto issue_tiles = [&](int k0, int b) {
        unsigned kb = (unsigned)k0 * 2u;
        asm volatile("global_load_async_to_lds_b128 %0, %1, %2"
                     :: "v"(ldsA0[b]), "v"(voffA0 + kb), "s"(Abase) : "memory");
        asm volatile("global_load_async_to_lds_b128 %0, %1, %2"
                     :: "v"(ldsA1[b]), "v"(voffA1 + kb), "s"(Abase) : "memory");
        asm volatile("global_load_async_to_lds_b128 %0, %1, %2"
                     :: "v"(ldsW0[b]), "v"(voffW0 + kb), "s"(Wbase) : "memory");
    };

    f32x8 acc[4] = {};
    issue_tiles(0, 0);
    int buf = 0;

    for (int k0 = 0; k0 < K; k0 += 32) {
        asm volatile("s_wait_asynccnt 0x0" ::: "memory");  // own tile DMA done
        __syncthreads();                                    // everyone's tile done
        if (k0 + 32 < K) issue_tiles(k0 + 32, buf ^ 1);     // prefetch next tile

        // --- A fragment (16x32): lane holds row m=l16; elems 0..7 = K[8*hi..],
        // elems 8..15 = K[16+8*hi..] (ISA 16-bit A layout)
        union { uint4 q[2]; bf16x16 v; } af;
        int arow = wave * 16 + l16;
        af.q[0] = *(const uint4*)(&As[buf][arow * 32 + hi * 8]);
        af.q[1] = *(const uint4*)(&As[buf][arow * 32 + 16 + hi * 8]);

#pragma unroll
        for (int t = 0; t < 4; ++t) {
            // --- B fragment (32x16): lane holds col n=l16, elems 0..7 = K[8*hi..], elems 8..15 = K[16+8*hi..]
            union { uint4 q[2]; bf16x16 v; } wf;
            int wrow = t * 16 + l16;
            wf.q[0] = *(const uint4*)(&Ws[buf][wrow * 32 + hi * 8]);
            wf.q[1] = *(const uint4*)(&Ws[buf][wrow * 32 + 16 + hi * 8]);
            acc[t] = __builtin_amdgcn_wmma_f32_16x16x32_bf16(
                false, af.v, false, wf.v, (short)0, acc[t], false, false);
        }
        buf ^= 1;
    }

    // --- store: VGPR r -> row (r + 8*hi), col l16 (ISA C/D layout)
    long rbase = m0 + wave * 16 + hi * 8;
#pragma unroll
    for (int t = 0; t < 4; ++t) {
        int col = n0 + t * 16 + l16;
#pragma unroll
        for (int r = 0; r < 8; ++r)
            C[(rbase + r) * (long)N + col] = acc[t][r];
    }
}

// ======================================================================
// Elementwise / staging kernels
// ======================================================================
__global__ void conv_w_pad(const float* __restrict__ w, unsigned short* __restrict__ out,
                           int Nv, int Kv, int Npad, int Kpad)
{
    long i = blockIdx.x * (long)blockDim.x + threadIdx.x;
    long total = (long)Npad * Kpad;
    if (i >= total) return;
    int k = (int)(i % Kpad); long n = i / Kpad;
    float v = (n < Nv && k < Kv) ? w[n * (long)Kv + k] : 0.f;
    out[i] = f2bf(v);
}

__global__ void f32_to_bf16_k(const float* __restrict__ in, unsigned short* __restrict__ out, long n)
{
    long i = blockIdx.x * (long)blockDim.x + threadIdx.x;
    if (i < n) out[i] = f2bf(in[i]);
}

// im2col for pe_conv1: rows (b,t,j), K = 3*9*17 = 459 padded to 480
__global__ void im2col1_k(const float* __restrict__ x, unsigned short* __restrict__ out)
{
    long i = blockIdx.x * (long)blockDim.x + threadIdx.x;
    long total = (long)MPE * 480;
    if (i >= total) return;
    int k = (int)(i % 480); long row = i / 480;
    float v = 0.f;
    if (row < NPOS && k < 459) {
        int kj = k % 17, kt = (k / 17) % 9, ci = k / 153;
        int j = (int)(row % 17), t = (int)((row / 17) % 243), b = (int)(row / 4131);
        int tt = t + kt - 4, jj = j + kj - 8;
        if (tt >= 0 && tt < TFRM && jj >= 0 && jj < JNT)
            v = x[(((long)b * TFRM + tt) * JNT + jj) * 3 + ci];
    }
    out[i] = f2bf(v);
}

// im2col for pe_conv2: rows (b,t,j), K = 192*3*17 = 9792
__global__ void im2col2_k(const float* __restrict__ h, unsigned short* __restrict__ out)
{
    long i = blockIdx.x * (long)blockDim.x + threadIdx.x;
    long total = (long)MPE * 9792;
    if (i >= total) return;
    int k = (int)(i % 9792); long row = i / 9792;
    float v = 0.f;
    if (row < NPOS) {
        int kj = k % 17, kt = (k / 17) % 3, ci = k / 51;
        int j = (int)(row % 17), t = (int)((row / 17) % 243), b = (int)(row / 4131);
        int tt = t + kt - 1, jj = j + kj - 8;
        if (tt >= 0 && tt < TFRM && jj >= 0 && jj < JNT)
            v = h[(((long)b * TFRM + tt) * JNT + jj) * 192 + ci];
    }
    out[i] = f2bf(v);
}

// bias + LayerNorm(192) + exact GELU, in-place (one block per position)
__global__ __launch_bounds__(256) void ln1_gelu_k(
    float* __restrict__ io, const float* __restrict__ cb,
    const float* __restrict__ lw, const float* __restrict__ lb)
{
    __shared__ float sm[256];
    int row = blockIdx.x, c = threadIdx.x;
    float v = (c < 192) ? io[(long)row * 192 + c] + cb[c] : 0.f;
    sm[c] = v; __syncthreads();
    for (int s = 128; s > 0; s >>= 1) { if (c < s) sm[c] += sm[c + s]; __syncthreads(); }
    float mean = sm[0] * (1.f / 192.f); __syncthreads();
    float d = (c < 192) ? (v - mean) : 0.f;
    sm[c] = d * d; __syncthreads();
    for (int s = 128; s > 0; s >>= 1) { if (c < s) sm[c] += sm[c + s]; __syncthreads(); }
    float var = sm[0] * (1.f / 192.f);
    if (c < 192) {
        float g = (v - mean) * rsqrtf(var + 1e-5f) * lw[c] + lb[c];
        io[(long)row * 192 + c] = 0.5f * g * (1.f + erff(g * 0.70710678118f));
    }
}

// bias + LayerNorm(384) + scatter into fwd/bwd token sequence
__global__ __launch_bounds__(512) void ln2_scatter_k(
    const float* __restrict__ in, const float* __restrict__ cb,
    const float* __restrict__ lw, const float* __restrict__ lb,
    float* __restrict__ seq)
{
    __shared__ float sm[512];
    int row = blockIdx.x, c = threadIdx.x;
    float v = (c < 384) ? in[(long)row * 384 + c] + cb[c] : 0.f;
    sm[c] = v; __syncthreads();
    for (int s = 256; s > 0; s >>= 1) { if (c < s) sm[c] += sm[c + s]; __syncthreads(); }
    float mean = sm[0] * (1.f / 384.f); __syncthreads();
    float d = (c < 384) ? (v - mean) : 0.f;
    sm[c] = d * d; __syncthreads();
    for (int s = 256; s > 0; s >>= 1) { if (c < s) sm[c] += sm[c + s]; __syncthreads(); }
    float var = sm[0] * (1.f / 384.f);
    if (c < 384) {
        float g = (v - mean) * rsqrtf(var + 1e-5f) * lw[c] + lb[c];
        int j = row % 17, t = (row / 17) % 243, b = row / 4131;
        int tf = (26 - t / 9) * 9 + t % 9;                 // per-patch flip (involution)
        long r1 = (long)b * LSEQ + t * 17 + j;
        long r2 = (long)b * LSEQ + 4131 + tf * 17 + j;
        seq[r1 * DMODEL + c] = g;
        seq[r2 * DMODEL + c] = g;
    }
}

// causal depthwise conv1d (width 4) + SiLU; writes f32 (for scan) and bf16 (for x_proj)
__global__ void dwconv_silu_k(const float* __restrict__ xz,
                              const float* __restrict__ cw, const float* __restrict__ cb,
                              float* __restrict__ xc, unsigned short* __restrict__ xcb)
{
    long i = blockIdx.x * (long)blockDim.x + threadIdx.x;
    if (i >= (long)NROWS * DINNER) return;
    int d = (int)(i % DINNER); long row = i / DINNER;
    int l = (int)(row % LSEQ);
    float acc = cb[d];
#pragma unroll
    for (int k = 0; k < 4; ++k) {
        int lk = l - 3 + k;
        if (lk >= 0) acc += xz[(row - 3 + k) * 1536 + d] * cw[d * 4 + k];
    }
    float s = acc / (1.f + expf(-acc));
    xc[i] = s;
    xcb[i] = f2bf(s);
}

// stage dbl[:, 0:24] into a zero-padded M x 32 bf16 operand for the dt GEMM
__global__ void dt_stage_k(const float* __restrict__ dbl, unsigned short* __restrict__ out)
{
    long i = blockIdx.x * (long)blockDim.x + threadIdx.x;
    if (i >= (long)MPAD * 32) return;
    int k = (int)(i % 32); long row = i / 32;
    out[i] = f2bf(k < 24 ? dbl[row * 64 + k] : 0.f);
}

__global__ void softplus_k(float* __restrict__ dt, const float* __restrict__ dtb)
{
    long i = blockIdx.x * (long)blockDim.x + threadIdx.x;
    if (i >= (long)NROWS * DINNER) return;
    float x = dt[i] + dtb[i % DINNER];
    dt[i] = (x > 20.f) ? x : log1pf(expf(x));
}

// selective scan: one thread per (batch, channel); 16-state recurrence in registers.
// Fuses the D skip-connection, SiLU(z) gate and bf16 conversion for out_proj.
__global__ __launch_bounds__(256) void scan_k(
    const float* __restrict__ dt, const float* __restrict__ dbl,
    const float* __restrict__ xc, const float* __restrict__ xz,
    const float* __restrict__ A_log, const float* __restrict__ Dv,
    unsigned short* __restrict__ yout)
{
    int b = blockIdx.x / 3;
    int d = (blockIdx.x % 3) * 256 + threadIdx.x;
    float A[DSTATE], h[DSTATE];
#pragma unroll
    for (int n = 0; n < DSTATE; ++n) { A[n] = -expf(A_log[d * DSTATE + n]); h[n] = 0.f; }
    float Dp = Dv[d];
    long base = (long)b * LSEQ;
    for (int l = 0; l < LSEQ; ++l) {
        long row = base + l;
        float dtv = dt[row * DINNER + d];
        float xv  = xc[row * DINNER + d];
        float zv  = xz[row * 1536 + DINNER + d];
        const float* bc = dbl + row * 64;
        float dx = dtv * xv, y = 0.f;
#pragma unroll
        for (int n = 0; n < DSTATE; ++n) {
            float dA = expf(dtv * A[n]);
            h[n] = dA * h[n] + dx * bc[24 + n];
            y += h[n] * bc[40 + n];
        }
        float yf = (y + xv * Dp) * (zv / (1.f + expf(-zv)));
        yout[row * DINNER + d] = f2bf(yf);
    }
}

// gather per-(b,t,j) 768-wide head feature: concat of token pair (l=2*(t*17+j)+{0,1})
__global__ void head_gather_k(const float* __restrict__ seq, unsigned short* __restrict__ out)
{
    long i = blockIdx.x * (long)blockDim.x + threadIdx.x;
    if (i >= (long)NPOS * 768) return;
    int c = (int)(i % 768); long row = i / 768;
    int j = (int)(row % 17), t = (int)((row / 17) % 243), b = (int)(row / 4131);
    int n = (c >= 384) ? 1 : 0, cc = (c >= 384) ? c - 384 : c;
    long l = ((long)t * 17 + j) * 2 + n;
    out[row * 768 + c] = f2bf(seq[((long)b * LSEQ + l) * DMODEL + cc]);
}

__global__ void tanh_stage_k(const float* __restrict__ h1, const float* __restrict__ b1,
                             unsigned short* __restrict__ out)
{
    long i = blockIdx.x * (long)blockDim.x + threadIdx.x;
    if (i >= (long)MPE * 384) return;
    out[i] = f2bf(tanhf(h1[i] + b1[i % 384]));
}

__global__ void final_k(const float* __restrict__ h2, const float* __restrict__ b2,
                        float* __restrict__ out)
{
    long i = blockIdx.x * (long)blockDim.x + threadIdx.x;
    if (i >= (long)NPOS * 3) return;
    int c = (int)(i % 3); long row = i / 3;
    out[i] = h2[row * 64 + c] + b2[c];
}

// ======================================================================
static inline int nblk(long n, int t) { return (int)((n + t - 1) / t); }

extern "C" void kernel_launch(void* const* d_in, const int* in_sizes, int n_in,
                              void* d_out, int out_size, void* d_ws, size_t ws_size,
                              hipStream_t stream)
{
    (void)in_sizes; (void)n_in; (void)out_size; (void)ws_size;
    const float* x        = (const float*)d_in[0];
    const float* pe1w     = (const float*)d_in[1];
    const float* pe1b     = (const float*)d_in[2];
    const float* ln1w     = (const float*)d_in[3];
    const float* ln1b     = (const float*)d_in[4];
    const float* pe2w     = (const float*)d_in[5];
    const float* pe2b     = (const float*)d_in[6];
    const float* ln2w     = (const float*)d_in[7];
    const float* ln2b     = (const float*)d_in[8];
    const float* m_in_w   = (const float*)d_in[9];
    const float* m_conv_w = (const float*)d_in[10];
    const float* m_conv_b = (const float*)d_in[11];
    const float* m_x_w    = (const float*)d_in[12];
    const float* m_dt_w   = (const float*)d_in[13];
    const float* m_dt_b   = (const float*)d_in[14];
    const float* m_A_log  = (const float*)d_in[15];
    const float* m_D      = (const float*)d_in[16];
    const float* m_out_w  = (const float*)d_in[17];
    const float* head_w1  = (const float*)d_in[18];
    const float* head_b1  = (const float*)d_in[19];
    const float* head_w2  = (const float*)d_in[20];
    const float* head_b2  = (const float*)d_in[21];
    float* out = (float*)d_out;

    char* ws = (char*)d_ws;
    size_t off = 0;
    auto alloc = [&](size_t bytes) -> char* {
        char* p = ws + off; off += (bytes + 255) & ~(size_t)255; return p;
    };
    float*          seq  = (float*)alloc((size_t)MPAD * 384 * 4);    //  50.9 MB
    float*          xz   = (float*)alloc((size_t)MPAD * 1536 * 4);   // 203.7 MB
    float*          xc   = (float*)alloc((size_t)MPAD * 768 * 4);    // 101.8 MB
    float*          dbl  = (float*)alloc((size_t)MPAD * 64 * 4);     //   8.5 MB
    float*          dtb  = (float*)alloc((size_t)MPAD * 768 * 4);    // 101.8 MB
    unsigned short* abuf = (unsigned short*)alloc((size_t)MPAD * 768 * 2); // 50.9 MB
    unsigned short* wbuf = (unsigned short*)alloc((size_t)384 * 9792 * 2); //  7.5 MB
    float*          pe1  = (float*)alloc((size_t)MPE * 192 * 4);     //  12.8 MB
    float*          pe2  = (float*)alloc((size_t)MPE * 384 * 4);     //  25.6 MB
    // im2col buffers alias the (not-yet-live) mamba activation region:
    unsigned short* ic1  = (unsigned short*)xz;                      // 16 MB
    unsigned short* ic2  = (unsigned short*)xz;                      // 326 MB, spans xz..dtb

    const int T = 256;

    // ---------------- patch embedding (both convs as WMMA implicit GEMM) ----
    im2col1_k<<<nblk((long)MPE * 480, T), T, 0, stream>>>(x, ic1);
    conv_w_pad<<<nblk((long)192 * 480, T), T, 0, stream>>>(pe1w, wbuf, 192, 459, 192, 480);
    gemm_wmma<<<dim3(192 / 64, MPE / 128), 256, 0, stream>>>(ic1, wbuf, pe1, 480, 192);
    ln1_gelu_k<<<NPOS, 256, 0, stream>>>(pe1, pe1b, ln1w, ln1b);

    im2col2_k<<<nblk((long)MPE * 9792, T), T, 0, stream>>>(pe1, ic2);
    conv_w_pad<<<nblk((long)384 * 9792, T), T, 0, stream>>>(pe2w, wbuf, 384, 9792, 384, 9792);
    gemm_wmma<<<dim3(384 / 64, MPE / 128), 256, 0, stream>>>(ic2, wbuf, pe2, 9792, 384);
    ln2_scatter_k<<<NPOS, 512, 0, stream>>>(pe2, pe2b, ln2w, ln2b, seq);

    // ---------------- 12 Mamba layers ----------------
    for (int i = 0; i < 12; ++i) {
        // in_proj: xz = seq @ in_w^T   (M x 384 -> M x 1536)
        conv_w_pad<<<nblk((long)1536 * 384, T), T, 0, stream>>>(
            m_in_w + (size_t)i * 1536 * 384, wbuf, 1536, 384, 1536, 384);
        f32_to_bf16_k<<<nblk((long)MPAD * 384, T), T, 0, stream>>>(seq, abuf, (long)MPAD * 384);
        gemm_wmma<<<dim3(1536 / 64, MPAD / 128), 256, 0, stream>>>(abuf, wbuf, xz, 384, 1536);

        // depthwise causal conv + SiLU (writes f32 + bf16 operand)
        dwconv_silu_k<<<nblk((long)NROWS * DINNER, T), T, 0, stream>>>(
            xz, m_conv_w + (size_t)i * DINNER * 4, m_conv_b + (size_t)i * DINNER, xc, abuf);

        // x_proj: dbl = xc @ x_w^T  (N 56 -> padded 64)
        conv_w_pad<<<nblk((long)64 * 768, T), T, 0, stream>>>(
            m_x_w + (size_t)i * 56 * 768, wbuf, 56, 768, 64, 768);
        gemm_wmma<<<dim3(1, MPAD / 128), 256, 0, stream>>>(abuf, wbuf, dbl, 768, 64);

        // dt_proj: dt = softplus(dbl[:, :24] @ dt_w^T + dt_b)  (K 24 -> padded 32)
        dt_stage_k<<<nblk((long)MPAD * 32, T), T, 0, stream>>>(dbl, abuf);
        conv_w_pad<<<nblk((long)768 * 32, T), T, 0, stream>>>(
            m_dt_w + (size_t)i * 768 * 24, wbuf, 768, 24, 768, 32);
        gemm_wmma<<<dim3(768 / 64, MPAD / 128), 256, 0, stream>>>(abuf, wbuf, dtb, 32, 768);
        softplus_k<<<nblk((long)NROWS * DINNER, T), T, 0, stream>>>(dtb, m_dt_b + (size_t)i * DINNER);

        // selective scan (fused D-skip + SiLU(z) gate), outputs bf16 operand
        scan_k<<<12, 256, 0, stream>>>(dtb, dbl, xc, xz,
                                       m_A_log + (size_t)i * DINNER * DSTATE,
                                       m_D + (size_t)i * DINNER, abuf);

        // out_proj: seq = y @ out_w^T
        conv_w_pad<<<nblk((long)384 * 768, T), T, 0, stream>>>(
            m_out_w + (size_t)i * 384 * 768, wbuf, 384, 768, 384, 768);
        gemm_wmma<<<dim3(384 / 64, MPAD / 128), 256, 0, stream>>>(abuf, wbuf, seq, 768, 384);
    }

    // ---------------- head ----------------
    head_gather_k<<<nblk((long)NPOS * 768, T), T, 0, stream>>>(seq, abuf);
    conv_w_pad<<<nblk((long)384 * 768, T), T, 0, stream>>>(head_w1, wbuf, 384, 768, 384, 768);
    gemm_wmma<<<dim3(384 / 64, MPE / 128), 256, 0, stream>>>(abuf, wbuf, pe2, 768, 384);
    tanh_stage_k<<<nblk((long)MPE * 384, T), T, 0, stream>>>(pe2, head_b1, abuf);
    conv_w_pad<<<nblk((long)64 * 384, T), T, 0, stream>>>(head_w2, wbuf, 3, 384, 64, 384);
    gemm_wmma<<<dim3(1, MPE / 128), 256, 0, stream>>>(abuf, wbuf, dbl, 384, 64);
    final_k<<<nblk((long)NPOS * 3, T), T, 0, stream>>>(dbl, head_b2, out);
}
